// DLSTM_13503377178714
// MI455X (gfx1250) — matrix operand order repaired and checked by reading, not compiled, not run on hardware
//
#include <hip/hip_runtime.h>

// ---------------- problem constants ----------------
constexpr int T_STEPS = 2048;
constexpr int BATCH   = 32;
constexpr int I_DIM   = 512;
constexpr int H_DIM   = 512;
constexpr int K1      = I_DIM + 2 * H_DIM;   // 1536 (W1 row stride)
constexpr int KR      = 2 * H_DIM;           // 1024 recurrent K (hx|cx)
constexpr int N1      = H_DIM;               // 512  -> 32 tiles of 16
constexpr int K2      = H_DIM;               // 512
constexpr int N2      = 5 * H_DIM;           // 2560 -> 160 tiles of 16
constexpr int NWG     = 40;                  // 40 WGs x 4 waves = 160 waves
constexpr int TPB     = 128;                 // 4 wave32 per WG
constexpr int SMEM_SCAN = BATCH * KR * 2;    // 64 KB (hx|cx panel)
constexpr int SMEM_PRE  = BATCH * I_DIM * 2; // 32 KB (x panel)

typedef __attribute__((ext_vector_type(16))) __bf16 v16bf;
typedef __attribute__((ext_vector_type(8)))  float  v8f;

union FragU { v16bf v; uint4 q[2]; };

// A-matrix 16x32 bf16 fragment (ISA 7.12.2): lane<16: M=lane, K {kk..kk+7},{kk+16..kk+23};
// lane>=16: K chunks +8.
__device__ __forceinline__ v16bf load_frag_a(const unsigned short* p) {
  FragU f;
  f.q[0] = *reinterpret_cast<const uint4*>(p);
  f.q[1] = *reinterpret_cast<const uint4*>(p + 16);
  return f.v;
}
// B-matrix 32x16 bf16 fragment: lane = N; lanes 0-15 K {0..7},{16..23}; lanes 16-31 K {8..15},{24..31}.
__device__ __forceinline__ v16bf load_frag_b(const unsigned short* p) {
  FragU f;
  f.q[0] = *reinterpret_cast<const uint4*>(p);
  f.q[1] = *reinterpret_cast<const uint4*>(p + 16);
  return f.v;
}

__device__ __forceinline__ unsigned short f2bf(float f) {
  unsigned u = __float_as_uint(f);
  u += 0x7FFFu + ((u >> 16) & 1u);                    // RNE
  return (unsigned short)(u >> 16);
}
__device__ __forceinline__ float sigmoidf(float x) {
  return 1.0f / (1.0f + __expf(-x));
}

// Software grid barrier: bar[0]=count, bar[1]=generation.
__device__ __forceinline__ void grid_barrier(unsigned* bar) {
  __threadfence();
  __syncthreads();
  if (threadIdx.x == 0) {
    unsigned gen = atomicAdd(&bar[1], 0u);
    if (atomicAdd(&bar[0], 1u) + 1u == (unsigned)gridDim.x) {
      atomicExch(&bar[0], 0u);
      __threadfence();
      atomicAdd(&bar[1], 1u);
    } else {
      while (atomicAdd(&bar[1], 0u) == gen) __builtin_amdgcn_s_sleep(8);
    }
  }
  __syncthreads();
  __threadfence();
}

// Copy a 32x512 bf16 row-major panel from global into LDS (uint4 rows of 64).
__device__ __forceinline__ void stage_panel(const unsigned short* __restrict__ src,
                                            uint4* __restrict__ sdst,
                                            int dstRowStrideU4, int dstColOffU4) {
  const uint4* s = reinterpret_cast<const uint4*>(src);
  for (int i = threadIdx.x; i < BATCH * 64; i += TPB) {
    int m = i >> 6, c = i & 63;
    sdst[m * dstRowStrideU4 + dstColOffU4 + c] = s[i];
  }
}

// ---------------- pre-pass kernels ----------------
__global__ void cvt_f32_to_bf16(const float* __restrict__ src,
                                unsigned short* __restrict__ dst, long n) {
  long i = (long)blockIdx.x * blockDim.x + threadIdx.x;
  long stride = (long)gridDim.x * blockDim.x;
  for (; i < n; i += stride) dst[i] = f2bf(src[i]);
}

__global__ void init_state(const float* __restrict__ mem,
                           unsigned short* __restrict__ hxbf,
                           unsigned short* __restrict__ cxbf,
                           float* __restrict__ cxf) {
  int i = blockIdx.x * blockDim.x + threadIdx.x;    // BATCH*H = 16384
  if (i < BATCH * H_DIM) {
    int m = i >> 9, j = i & (H_DIM - 1);
    float hx = mem[m * (2 * H_DIM) + j];
    float cx = mem[m * (2 * H_DIM) + H_DIM + j];
    hxbf[i] = f2bf(hx);
    cxbf[i] = f2bf(cx);
    cxf[i]  = cx;
  }
}

// Z = X @ W1x^T for all timesteps (input contribution of GEMM1, recurrence-free).
// One WG per 32-row M-group of X (T*B/32 = 2048 WGs); waves cover the 32 N-tiles.
__global__ __launch_bounds__(TPB)
void precompute_zx(const unsigned short* __restrict__ xflat,  // (T*B, 512) bf16
                   const unsigned short* __restrict__ w1,     // (512, 1536) bf16
                   float* __restrict__ z) {                   // (T*B, 512) f32
  extern __shared__ unsigned char smem_raw[];
  unsigned short* sA  = reinterpret_cast<unsigned short*>(smem_raw);
  uint4*          sU4 = reinterpret_cast<uint4*>(smem_raw);

  const int lane  = threadIdx.x & 31;
  const int wave  = threadIdx.x >> 5;
  const int l15   = lane & 15;
  const int khalf = lane >> 4;
  const int aoff  = khalf * 8;
  const int boff  = khalf * 8;

  const size_t mbase = (size_t)blockIdx.x * BATCH;            // first row of M-group
  stage_panel(xflat + mbase * I_DIM, sU4, 64, 0);             // 32x512 panel
  __syncthreads();

  for (int nt = wave; nt < N1 / 16; nt += (TPB >> 5)) {
    const int n0 = nt * 16;
    const unsigned short* wrow = w1 + (size_t)(n0 + l15) * K1;   // cols 0..511
    const unsigned short* aLoP = sA + l15 * I_DIM + aoff;
    const unsigned short* aHiP = sA + (16 + l15) * I_DIM + aoff;

    v8f cLo = {}, cHi = {};
    v16bf bf0 = load_frag_b(wrow + boff);
    v16bf bf1 = load_frag_b(wrow + 32 + boff);
    v16bf aL0 = load_frag_a(aLoP);
    v16bf aH0 = load_frag_a(aHiP);
    for (int k = 32; k < I_DIM; k += 32) {
      v16bf bf2 = load_frag_b(wrow + k + 32 + boff);   // in-row over-read ok (K1>512+64)
      v16bf aL1 = load_frag_a(aLoP + k);
      v16bf aH1 = load_frag_a(aHiP + k);
      __builtin_prefetch(wrow + k + 96, 0, 1);
      cLo = __builtin_amdgcn_wmma_f32_16x16x32_bf16(false, aL0, false, bf0,
                                                    (short)0, cLo, false, false);
      cHi = __builtin_amdgcn_wmma_f32_16x16x32_bf16(false, aH0, false, bf0,
                                                    (short)0, cHi, false, false);
      bf0 = bf1; bf1 = bf2; aL0 = aL1; aH0 = aH1;
    }
    cLo = __builtin_amdgcn_wmma_f32_16x16x32_bf16(false, aL0, false, bf0,
                                                  (short)0, cLo, false, false);
    cHi = __builtin_amdgcn_wmma_f32_16x16x32_bf16(false, aH0, false, bf0,
                                                  (short)0, cHi, false, false);
    #pragma unroll
    for (int v = 0; v < 8; ++v) {
      z[(mbase + khalf * 8 + v)      * N1 + n0 + l15] = cLo[v];
      z[(mbase + 16 + khalf * 8 + v) * N1 + n0 + l15] = cHi[v];
    }
  }
}

// ---------------- persistent scan kernel ----------------
__global__ __launch_bounds__(TPB)
void dlstm_scan(const float* __restrict__ z,                // (T*B, 512) f32 precomputed
                const float* __restrict__ rst,              // (T,B)
                const unsigned short* __restrict__ w1,      // (N1,K1) bf16
                const float* __restrict__ b1,               // (N1)
                const unsigned short* __restrict__ w2,      // (N2,K2) bf16
                const float* __restrict__ b2,               // (N2)
                unsigned short* __restrict__ hxbf,          // (B,H) bf16 state
                unsigned short* __restrict__ cxbf,          // (B,H) bf16 state
                float* __restrict__ cxf,                    // (B,H) f32 state
                unsigned short* __restrict__ hbf,           // (B,H) bf16 scratch
                float* __restrict__ net,                    // (B,N2) f32 scratch
                float* __restrict__ out,                    // (T,B,H) + (B,2H)
                unsigned* __restrict__ bar) {
  extern __shared__ unsigned char smem_raw[];
  unsigned short* sA  = reinterpret_cast<unsigned short*>(smem_raw);
  uint4*          sU4 = reinterpret_cast<uint4*>(smem_raw);

  const int lane  = threadIdx.x & 31;
  const int wave  = threadIdx.x >> 5;
  const int gw    = blockIdx.x * (TPB >> 5) + wave;        // 0..159
  const int l15   = lane & 15;
  const int khalf = lane >> 4;
  const int aoff  = khalf * 8;
  const int boff  = khalf * 8;

  for (int t = 0; t < T_STEPS; ++t) {
    // ---- GEMM1 (recurrent part): h = tanh(Z_t + [hx|cx] @ W1r^T + b1) ----
    if (blockIdx.x < (N1 / 16) / (TPB >> 5)) {             // WGs 0..7
      stage_panel(hxbf, sU4, 128, 0);                      // 32 x 1024 panel
      stage_panel(cxbf, sU4, 128, 64);
      __syncthreads();

      const int n0 = gw * 16;
      const unsigned short* wrow = w1 + (size_t)(n0 + l15) * K1 + I_DIM;  // cols 512..1535
      const unsigned short* aLoP = sA + l15 * KR + aoff;
      const unsigned short* aHiP = sA + (16 + l15) * KR + aoff;
      const float* zrow = z + (size_t)t * BATCH * N1;

      v8f cLo, cHi;
      #pragma unroll
      for (int v = 0; v < 8; ++v) {                        // seed from precomputed Z
        cLo[v] = zrow[(khalf * 8 + v)      * N1 + n0 + l15];
        cHi[v] = zrow[(16 + khalf * 8 + v) * N1 + n0 + l15];
      }
      v16bf bf0 = load_frag_b(wrow + boff);
      v16bf bf1 = load_frag_b(wrow + 32 + boff);
      v16bf aL0 = load_frag_a(aLoP);
      v16bf aH0 = load_frag_a(aHiP);
      for (int k = 32; k < KR; k += 32) {
        v16bf bf2 = load_frag_b(wrow + k + 32 + boff);     // tail over-read stays in ws
        v16bf aL1 = load_frag_a(aLoP + k);
        v16bf aH1 = load_frag_a(aHiP + k);
        __builtin_prefetch(wrow + k + 96, 0, 1);
        cLo = __builtin_amdgcn_wmma_f32_16x16x32_bf16(false, aL0, false, bf0,
                                                      (short)0, cLo, false, false);
        cHi = __builtin_amdgcn_wmma_f32_16x16x32_bf16(false, aH0, false, bf0,
                                                      (short)0, cHi, false, false);
        bf0 = bf1; bf1 = bf2; aL0 = aL1; aH0 = aH1;
      }
      cLo = __builtin_amdgcn_wmma_f32_16x16x32_bf16(false, aL0, false, bf0,
                                                    (short)0, cLo, false, false);
      cHi = __builtin_amdgcn_wmma_f32_16x16x32_bf16(false, aH0, false, bf0,
                                                    (short)0, cHi, false, false);

      const float bias = b1[n0 + l15];
      #pragma unroll
      for (int v = 0; v < 8; ++v) {
        const int mLo = khalf * 8 + v;
        const int mHi = 16 + khalf * 8 + v;
        hbf[mLo * N1 + n0 + l15] = f2bf(tanhf(cLo[v] + bias));
        hbf[mHi * N1 + n0 + l15] = f2bf(tanhf(cHi[v] + bias));
      }
    }
    grid_barrier(bar);

    // ---- GEMM2: net = h @ W2^T + b2; 160 waves, one 32x16 tile each ----
    {
      stage_panel(hbf, sU4, 64, 0);                        // 32 x 512 h panel
      __syncthreads();

      const int n0 = gw * 16;
      const unsigned short* wrow = w2 + (size_t)(n0 + l15) * K2;
      const unsigned short* aLoP = sA + l15 * K2 + aoff;
      const unsigned short* aHiP = sA + (16 + l15) * K2 + aoff;

      v8f cLo = {}, cHi = {};
      v16bf bf0 = load_frag_b(wrow + boff);
      v16bf bf1 = load_frag_b(wrow + 32 + boff);
      v16bf aL0 = load_frag_a(aLoP);
      v16bf aH0 = load_frag_a(aHiP);
      for (int k = 32; k < K2; k += 32) {
        v16bf bf2 = load_frag_b(wrow + k + 32 + boff);
        v16bf aL1 = load_frag_a(aLoP + k);
        v16bf aH1 = load_frag_a(aHiP + k);
        __builtin_prefetch(wrow + k + 96, 0, 1);
        cLo = __builtin_amdgcn_wmma_f32_16x16x32_bf16(false, aL0, false, bf0,
                                                      (short)0, cLo, false, false);
        cHi = __builtin_amdgcn_wmma_f32_16x16x32_bf16(false, aH0, false, bf0,
                                                      (short)0, cHi, false, false);
        bf0 = bf1; bf1 = bf2; aL0 = aL1; aH0 = aH1;
      }
      cLo = __builtin_amdgcn_wmma_f32_16x16x32_bf16(false, aL0, false, bf0,
                                                    (short)0, cLo, false, false);
      cHi = __builtin_amdgcn_wmma_f32_16x16x32_bf16(false, aH0, false, bf0,
                                                    (short)0, cHi, false, false);

      const float bias = b2[n0 + l15];
      #pragma unroll
      for (int v = 0; v < 8; ++v) {
        const int mLo = khalf * 8 + v;
        const int mHi = 16 + khalf * 8 + v;
        net[mLo * N2 + n0 + l15] = cLo[v] + bias;
        net[mHi * N2 + n0 + l15] = cHi[v] + bias;
      }
    }
    grid_barrier(bar);

    // ---- elementwise gates + state update ----
    const int gtid = blockIdx.x * TPB + threadIdx.x;
    const int nthr = gridDim.x * TPB;
    for (int idx = gtid; idx < BATCH * H_DIM; idx += nthr) {
      const int m = idx >> 9, j = idx & (H_DIM - 1);
      const float* nrow = net + m * N2;
      const float hx_pre = nrow[j];
      const float ncx    = nrow[H_DIM + j];
      const float og     = sigmoidf(nrow[2 * H_DIM + j]);
      float       fg     = sigmoidf(nrow[3 * H_DIM + j]);
      float       ig     = sigmoidf(nrow[4 * H_DIM + j]);
      const float rt     = rst[t * BATCH + m];
      ig *= (1.0f - rt);
      fg *= rt;
      const float hnew = tanhf(hx_pre) * og;
      const float cnew = ig * tanhf(ncx) + fg * cxf[idx];
      out[(size_t)t * (BATCH * H_DIM) + idx] = hnew;
      hxbf[idx] = f2bf(hnew);
      cxf[idx]  = cnew;
      cxbf[idx] = f2bf(cnew);
      if (t == T_STEPS - 1) {
        float* memout = out + (size_t)T_STEPS * BATCH * H_DIM;
        memout[m * 2 * H_DIM + j]         = hnew;
        memout[m * 2 * H_DIM + H_DIM + j] = cnew;
      }
    }
    grid_barrier(bar);
  }
}

// ---------------- host launcher ----------------
extern "C" void kernel_launch(void* const* d_in, const int* in_sizes, int n_in,
                              void* d_out, int out_size, void* d_ws, size_t ws_size,
                              hipStream_t stream) {
  const float* inp = (const float*)d_in[0];   // (T,B,I)
  const float* mem = (const float*)d_in[1];   // (1,B,2H)
  const float* rst = (const float*)d_in[2];   // (T,B)
  const float* W1  = (const float*)d_in[3];   // (H, 2H+I)
  const float* b1  = (const float*)d_in[4];   // (H)
  const float* W2  = (const float*)d_in[5];   // (5H, H)
  const float* b2  = (const float*)d_in[6];   // (5H)
  float* out = (float*)d_out;

  char* ws = (char*)d_ws;
  size_t off = 0;
  auto take = [&](size_t bytes) -> char* {
    char* p = ws + off;
    off = (off + bytes + 255) & ~(size_t)255;
    return p;
  };
  unsigned short* xbf  = (unsigned short*)take((size_t)T_STEPS * BATCH * I_DIM * 2);
  unsigned short* w1bf = (unsigned short*)take((size_t)N1 * K1 * 2);
  unsigned short* w2bf = (unsigned short*)take((size_t)N2 * K2 * 2);
  unsigned short* hxbf = (unsigned short*)take((size_t)BATCH * H_DIM * 2);
  unsigned short* cxbf = (unsigned short*)take((size_t)BATCH * H_DIM * 2);
  float*          cxf  = (float*)take((size_t)BATCH * H_DIM * 4);
  unsigned short* hbf  = (unsigned short*)take((size_t)BATCH * H_DIM * 2);
  float*          net  = (float*)take((size_t)BATCH * N2 * 4);
  unsigned*       bar  = (unsigned*)take(256);
  float*          zbuf = (float*)take((size_t)T_STEPS * BATCH * N1 * 4);  // 128 MB

  hipMemsetAsync(bar, 0, 256, stream);

  const long nX  = (long)T_STEPS * BATCH * I_DIM;
  const long nW1 = (long)N1 * K1;
  const long nW2 = (long)N2 * K2;
  unsigned bX  = (unsigned)((nX  + 1023) / 1024);
  unsigned bW1 = (unsigned)((nW1 + 1023) / 1024);
  unsigned bW2 = (unsigned)((nW2 + 1023) / 1024);
  cvt_f32_to_bf16<<<bX,  256, 0, stream>>>(inp, xbf,  nX);
  cvt_f32_to_bf16<<<bW1, 256, 0, stream>>>(W1,  w1bf, nW1);
  cvt_f32_to_bf16<<<bW2, 256, 0, stream>>>(W2,  w2bf, nW2);
  init_state<<<(BATCH * H_DIM + 255) / 256, 256, 0, stream>>>(mem, hxbf, cxbf, cxf);

  // Recurrence-free input projection for all timesteps, full-device WMMA GEMM.
  precompute_zx<<<(T_STEPS * BATCH) / BATCH, TPB, SMEM_PRE, stream>>>(xbf, w1bf, zbuf);

  dlstm_scan<<<NWG, TPB, SMEM_SCAN, stream>>>(zbuf, rst, w1bf, b1, w2bf, b2,
                                              hxbf, cxbf, cxf, hbf, net, out, bar);
}
